// AttentionModel_47725676593770
// MI455X (gfx1250) — matrix operand order repaired and checked by reading, not compiled, not run on hardware
//
#include <hip/hip_runtime.h>
#include <math.h>

// ---------------- problem constants ----------------
#define NFFT    2048
#define HOPL    512
#define NBINS   1025
#define DM      2050        // d_model
#define NH      2
#define HD      1025        // head dim
#define NB      4           // batch
#define TLEN    262144
#define SFR     513         // frames = T/HOP + 1
#define DFF     8200        // 4*DM
#define LOLA    (HOPL*(SFR-1)+NFFT)   // 264192
#define LNEPS   1e-5f

typedef __attribute__((ext_vector_type(16))) _Float16 v16h;
typedef __attribute__((ext_vector_type(8)))  _Float16 v8h;
typedef __attribute__((ext_vector_type(8)))  float    v8f;

union Frag16 { v16h v; v8h h[2]; };

// ---- CDNA5 async global->LDS copy helpers (ASYNCcnt-tracked DMA path) ----
// GVS form: mem_addr = SADDR(64) + VADDR(u32 byte offset); VDST = LDS byte addr.
// Generic LDS pointers carry the LDS address in bits [31:0] (aperture in [63:32]).
__device__ __forceinline__ void async_copy_b32(const float* gbase, unsigned byteOff,
                                               float* ldsDst)
{
    unsigned lds = (unsigned)(size_t)ldsDst;
    asm volatile("global_load_async_to_lds_b32 %0, %1, %2"
                 :: "v"(lds), "v"(byteOff), "s"(gbase) : "memory");
}
__device__ __forceinline__ void async_wait_all()
{
    asm volatile("s_wait_asynccnt 0" ::: "memory");
}

// =====================================================================
// Generic batched GEMM:  C = alpha*op(A,B) + biasScale*bias[n] (+C) (relu)
//   A: (M,K) row-major, lda, batch stride sA
//   B: opBT=1 -> (N,K) row-major (i.e. compute A*B^T), else (K,N) row-major
//   C: (M,N) row-major
// f32 -> f16 convert in LDS staging, f32 accumulate via v_wmma_f32_16x16x32_f16
// =====================================================================
#define BM 128
#define BN 128
#define BKK 32
#define LDST 40   // BKK + 8 pad (80B row stride: conflict-spread, 16B aligned)

__global__ __launch_bounds__(256) void gemm_f16_wmma(
    const float* __restrict__ A, long long lda, long long sA,
    const float* __restrict__ B, long long ldb, long long sB, int opBT,
    float* __restrict__ C, long long ldc, long long sC,
    int M, int N, int K, float alpha,
    const float* __restrict__ bias, float biasScale, int accFlag, int reluFlag)
{
    __shared__ __attribute__((aligned(16))) _Float16 As[2][BM * LDST];
    __shared__ __attribute__((aligned(16))) _Float16 Bs[2][BN * LDST];

    const int tid  = threadIdx.x;
    const int lane = tid & 31;
    const int wave = tid >> 5;          // 8 waves
    const int wr   = wave & 3;          // wave row (0..3)  -> 32 rows each
    const int wc   = wave >> 2;         // wave col (0..1)  -> 64 cols each
    const int half = lane >> 4;         // 0/1 (wave32 half)
    const int l16  = lane & 15;

    const int bz    = blockIdx.z;
    const int tileM = blockIdx.y * BM;
    const int tileN = blockIdx.x * BN;

    const float* Ab = A + (size_t)bz * sA;
    const float* Bb = B + (size_t)bz * sB;
    float*       Cb = C + (size_t)bz * sC;

    float ra[16], rb[16];

    auto stageLoad = [&](int k0) {
#pragma unroll
        for (int i = 0; i < 4; ++i) {
            int slot = tid + i * 256;       // 0..1023
            int row  = slot >> 3;           // 0..127
            int kq   = (slot & 7) << 2;     // 0,4,...,28
            int gm = tileM + row;
            int gn = tileN + row;
#pragma unroll
            for (int j = 0; j < 4; ++j) {
                int gk = k0 + kq + j;
                float av = 0.f, bv = 0.f;
                if (gm < M && gk < K) av = Ab[(size_t)gm * lda + gk];
                if (gn < N && gk < K)
                    bv = opBT ? Bb[(size_t)gn * ldb + gk]
                              : Bb[(size_t)gk * ldb + gn];
                ra[i * 4 + j] = av;
                rb[i * 4 + j] = bv;
            }
        }
    };
    auto stageStore = [&](int buf) {
#pragma unroll
        for (int i = 0; i < 4; ++i) {
            int slot = tid + i * 256;
            int row  = slot >> 3;
            int kq   = (slot & 7) << 2;
#pragma unroll
            for (int j = 0; j < 4; ++j) {
                As[buf][row * LDST + kq + j] = (_Float16)ra[i * 4 + j];
                Bs[buf][row * LDST + kq + j] = (_Float16)rb[i * 4 + j];
            }
        }
    };

    v8f zero = {0.f,0.f,0.f,0.f,0.f,0.f,0.f,0.f};
    v8f acc[2][4];
#pragma unroll
    for (int mt = 0; mt < 2; ++mt)
#pragma unroll
        for (int nt = 0; nt < 4; ++nt) acc[mt][nt] = zero;

    const int nK = (K + BKK - 1) / BKK;

    stageLoad(0);
    stageStore(0);
    __syncthreads();

    for (int kt = 0; kt < nK; ++kt) {
        const int cur = kt & 1;
        if (kt + 1 < nK) stageLoad((kt + 1) * BKK);
        if (kt + 2 < nK) {  // speculative L2 prefetch (global_prefetch_b8)
            int pr = tid >> 3;
            __builtin_prefetch(&Ab[(size_t)(tileM + pr) * lda + (size_t)(kt + 2) * BKK], 0, 1);
            __builtin_prefetch(opBT ? &Bb[(size_t)(tileN + pr) * ldb + (size_t)(kt + 2) * BKK]
                                    : &Bb[(size_t)((kt + 2) * BKK) * ldb + tileN + pr], 0, 1);
        }

        // fragments for this k-tile
        Frag16 af[2], bf[4];
#pragma unroll
        for (int mt = 0; mt < 2; ++mt) {
            int row = wr * 32 + mt * 16 + l16;
            const v8h* p0 = (const v8h*)&As[cur][row * LDST + half * 8];
            const v8h* p1 = (const v8h*)&As[cur][row * LDST + half * 8 + 16];
            af[mt].h[0] = *p0;              // K = klo .. klo+7
            af[mt].h[1] = *p1;              // K = klo+16 .. klo+23
        }
#pragma unroll
        for (int nt = 0; nt < 4; ++nt) {
            int col = wc * 64 + nt * 16 + l16;
            const v8h* q = (const v8h*)&Bs[cur][col * LDST + half * 8];
            bf[nt].h[0] = q[0];             // K = kb .. kb+7
            bf[nt].h[1] = q[2];             // K = kb+16 .. kb+23
        }
#pragma unroll
        for (int mt = 0; mt < 2; ++mt)
#pragma unroll
            for (int nt = 0; nt < 4; ++nt)
                acc[mt][nt] = __builtin_amdgcn_wmma_f32_16x16x32_f16(
                    false, af[mt].v, false, bf[nt].v,
                    (short)0, acc[mt][nt], false, false);

        if (kt + 1 < nK) stageStore(cur ^ 1);
        __syncthreads();
    }

    // epilogue: C layout — lane l: n = base + l%16 ; m = base + vgpr + 8*(l/16)
#pragma unroll
    for (int nt = 0; nt < 4; ++nt) {
        int n = tileN + wc * 64 + nt * 16 + l16;
        if (n >= N) continue;
        float bvl = bias ? biasScale * bias[n] : 0.f;
#pragma unroll
        for (int mt = 0; mt < 2; ++mt) {
#pragma unroll
            for (int i = 0; i < 8; ++i) {
                int m = tileM + wr * 32 + mt * 16 + half * 8 + i;
                if (m >= M) continue;
                size_t off = (size_t)m * ldc + n;
                float v = alpha * acc[mt][nt][i] + bvl;
                if (accFlag) v += Cb[off];
                if (reluFlag) v = v > 0.f ? v : 0.f;
                Cb[off] = v;
            }
        }
    }
}

static inline void launch_gemm(hipStream_t st,
    const float* A, long long lda, long long sA,
    const float* B, long long ldb, long long sB, int opBT,
    float* C, long long ldc, long long sC,
    int M, int N, int K, int batch,
    float alpha, const float* bias, float biasScale, int acc, int relu)
{
    dim3 grid((N + BN - 1) / BN, (M + BM - 1) / BM, batch);
    gemm_f16_wmma<<<grid, dim3(256), 0, st>>>(A, lda, sA, B, ldb, sB, opBT,
                                              C, ldc, sC, M, N, K, alpha,
                                              bias, biasScale, acc, relu);
}

// =====================================================================
// twiddle tables + window-sum
// =====================================================================
__global__ void tw_kernel(float* ct, float* st) {
    int i = blockIdx.x * 256 + threadIdx.x;
    if (i < NFFT) {
        float a = 6.2831853071795864f * (float)i / (float)NFFT;
        ct[i] = cosf(a);
        st[i] = sinf(a);
    }
}

__global__ void wsum_kernel(const float* __restrict__ win, float* __restrict__ wsum) {
    int t = blockIdx.x * 256 + threadIdx.x;
    if (t >= LOLA) return;
    int shi = t / HOPL; if (shi > SFR - 1) shi = SFR - 1;
    int slo = (t >= NFFT) ? ((t - NFFT) / HOPL + 1) : 0;
    float acc = 0.f;
    for (int s = slo; s <= shi; ++s) {
        float w = win[t - s * HOPL];
        acc += w * w;
    }
    wsum[t] = acc;
}

// =====================================================================
// STFT (table DFT) + positional encoding, writes xr/xi in (b,s,d) layout
// Twiddle tables are brought into LDS with async global->LDS DMA, overlapped
// with the windowed-frame construction.
// =====================================================================
__global__ __launch_bounds__(256) void stft_kernel(
    const float* __restrict__ mix, const float* __restrict__ win,
    const float* __restrict__ ct, const float* __restrict__ st,
    float* __restrict__ xr, float* __restrict__ xi)
{
    __shared__ float fx[NFFT];
    __shared__ float sct[NFFT];
    __shared__ float sst[NFFT];
    const int s = blockIdx.x;       // frame
    const int g = blockIdx.y;       // signal 0..7
    const int tid = threadIdx.x;

    // issue async table copies first (ASYNCcnt), overlap with frame build
    for (int n = tid; n < NFFT; n += 256) {
        async_copy_b32(ct, (unsigned)(n * 4), &sct[n]);
        async_copy_b32(st, (unsigned)(n * 4), &sst[n]);
    }
    for (int n = tid; n < NFFT; n += 256) {
        int j = s * HOPL + n - NFFT / 2;     // reflect pad
        if (j < 0) j = -j;
        if (j >= TLEN) j = 2 * TLEN - 2 - j;
        fx[n] = mix[(size_t)g * TLEN + j] * win[n];
    }
    async_wait_all();
    __syncthreads();

    const int bseq = g >> 1, c = g & 1;
    for (int b = tid; b < NBINS; b += 256) {
        float are = 0.f, aim = 0.f;
        for (int n = 0; n < NFFT; ++n) {
            int idx = (b * n) & (NFFT - 1);
            float x = fx[n];
            are += x * sct[idx];
            aim -= x * sst[idx];
        }
        // positional encoding: i = b ; ang = s * exp(-2b*ln(1e4)/D)
        float div = expf(-(float)(2 * b) * (9.2103403719761836f / (float)DM));
        float ang = (float)s * div;
        float enc = (c == 0) ? sinf(ang) : cosf(ang);
        size_t off = ((size_t)(bseq * SFR + s)) * DM + 2 * b + c;
        xr[off] = are + enc;
        xi[off] = aim + enc;
    }
}

// =====================================================================
// softmax over rows of length SFR (in place)
// =====================================================================
__global__ __launch_bounds__(256) void softmax_kernel(float* __restrict__ x) {
    __shared__ float buf[SFR];
    __shared__ float red[256];
    const int tid = threadIdx.x;
    float* p = x + (size_t)blockIdx.x * SFR;

    float mx = -3.4e38f;
    for (int i = tid; i < SFR; i += 256) { float v = p[i]; buf[i] = v; mx = fmaxf(mx, v); }
    red[tid] = mx; __syncthreads();
    for (int s = 128; s > 0; s >>= 1) { if (tid < s) red[tid] = fmaxf(red[tid], red[tid + s]); __syncthreads(); }
    mx = red[0]; __syncthreads();

    float sum = 0.f;
    for (int i = tid; i < SFR; i += 256) { float e = expf(buf[i] - mx); buf[i] = e; sum += e; }
    red[tid] = sum; __syncthreads();
    for (int s = 128; s > 0; s >>= 1) { if (tid < s) red[tid] += red[tid + s]; __syncthreads(); }
    float inv = 1.0f / red[0];
    for (int i = tid; i < SFR; i += 256) p[i] = buf[i] * inv;
}

// =====================================================================
// LayerNorm over D, optional residual:  y = g*norm(x (+res)) + b
// =====================================================================
__global__ __launch_bounds__(256) void ln_kernel(
    const float* __restrict__ x, const float* __restrict__ res,
    const float* __restrict__ g, const float* __restrict__ b,
    float* __restrict__ y)
{
    __shared__ float buf[DM];
    __shared__ float red[256];
    const int tid = threadIdx.x;
    const size_t row = blockIdx.x;
    const float* px = x + row * DM;
    const float* pr = res ? res + row * DM : nullptr;

    float sum = 0.f;
    for (int i = tid; i < DM; i += 256) {
        float v = px[i] + (pr ? pr[i] : 0.f);
        buf[i] = v; sum += v;
    }
    red[tid] = sum; __syncthreads();
    for (int s = 128; s > 0; s >>= 1) { if (tid < s) red[tid] += red[tid + s]; __syncthreads(); }
    float mu = red[0] / (float)DM; __syncthreads();

    float sq = 0.f;
    for (int i = tid; i < DM; i += 256) { float d = buf[i] - mu; sq += d * d; }
    red[tid] = sq; __syncthreads();
    for (int s = 128; s > 0; s >>= 1) { if (tid < s) red[tid] += red[tid + s]; __syncthreads(); }
    float rs = rsqrtf(red[0] / (float)DM + LNEPS);

    float* py = y + row * DM;
    for (int i = tid; i < DM; i += 256)
        py[i] = (buf[i] - mu) * rs * g[i] + b[i];
}

// =====================================================================
// sigmoid gate (in place into nr2/ni2) + scatter the zc output block
// out2 layout: (8, BINS, S, 2)  [re, im]
// =====================================================================
__global__ void gate_scatter_kernel(
    const float* __restrict__ xr, const float* __restrict__ xi,
    float* __restrict__ nr2, float* __restrict__ ni2, float* __restrict__ out2)
{
    size_t idx = (size_t)blockIdx.x * 256 + threadIdx.x;
    const size_t NTOT = (size_t)NB * SFR * DM;
    if (idx >= NTOT) return;
    size_t bd = idx / DM;
    int d = (int)(idx - bd * DM);
    int s = (int)(bd % SFR);
    int b = (int)(bd / SFR);
    float gr = xr[idx] * (1.0f / (1.0f + expf(-nr2[idx])));
    float gi = xi[idx] * (1.0f / (1.0f + expf(-ni2[idx])));
    nr2[idx] = gr;
    ni2[idx] = gi;
    int bin = d >> 1, c = d & 1, gsig = b * 2 + c;
    size_t o = (((size_t)gsig * NBINS + bin) * SFR + s) * 2;
    out2[o]     = gr;
    out2[o + 1] = gi;
}

// =====================================================================
// ISTFT frames: iDFT per (signal, frame), fused window multiply.
// Spectrum row and twiddle tables fetched with async global->LDS DMA.
// =====================================================================
__global__ __launch_bounds__(256) void istft_frames_kernel(
    const float* __restrict__ gr, const float* __restrict__ gi,
    const float* __restrict__ win,
    const float* __restrict__ ct, const float* __restrict__ st,
    float* __restrict__ frames)
{
    __shared__ float re[NBINS];
    __shared__ float im[NBINS];
    __shared__ float sct[NFFT];
    __shared__ float sst[NFFT];
    const int s = blockIdx.x;
    const int g = blockIdx.y;
    const int tid = threadIdx.x;
    const int bseq = g >> 1, c = g & 1;

    for (int k = tid; k < NBINS; k += 256) {
        unsigned goff = (unsigned)(((((size_t)(bseq * SFR + s)) * DM) + 2 * k + c) * 4);
        async_copy_b32(gr, goff, &re[k]);
        async_copy_b32(gi, goff, &im[k]);
    }
    for (int n = tid; n < NFFT; n += 256) {
        async_copy_b32(ct, (unsigned)(n * 4), &sct[n]);
        async_copy_b32(st, (unsigned)(n * 4), &sst[n]);
    }
    async_wait_all();
    __syncthreads();

    for (int n = tid; n < NFFT; n += 256) {
        float acc = 0.f;
        for (int k = 0; k <= NFFT / 2; ++k) {
            float w = (k == 0 || k == NFFT / 2) ? 1.0f : 2.0f;
            int idx = (k * n) & (NFFT - 1);
            acc += w * (re[k] * sct[idx] - im[k] * sst[idx]);
        }
        frames[((size_t)g * SFR + s) * NFFT + n] = acc * (1.0f / (float)NFFT) * win[n];
    }
}

// =====================================================================
// overlap-add + wsum normalization; writes estim (8, T)
// =====================================================================
__global__ void ola_kernel(const float* __restrict__ frames,
                           const float* __restrict__ wsum,
                           float* __restrict__ out1)
{
    size_t idx = (size_t)blockIdx.x * 256 + threadIdx.x;
    const size_t NTOT = (size_t)8 * TLEN;
    if (idx >= NTOT) return;
    int g = (int)(idx / TLEN);
    int t = (int)(idx - (size_t)g * TLEN);
    int tp = t + NFFT / 2;
    int shi = tp / HOPL; if (shi > SFR - 1) shi = SFR - 1;
    int slo = (tp >= NFFT) ? ((tp - NFFT) / HOPL + 1) : 0;
    float acc = 0.f;
    for (int s = slo; s <= shi; ++s)
        acc += frames[((size_t)g * SFR + s) * NFFT + (tp - s * HOPL)];
    out1[idx] = acc / wsum[tp];
}

// =====================================================================
// orchestration
// =====================================================================
extern "C" void kernel_launch(void* const* d_in, const int* in_sizes, int n_in,
                              void* d_out, int out_size, void* d_ws, size_t ws_size,
                              hipStream_t stream)
{
    const float* mix    = (const float*)d_in[0];
    const float* window = (const float*)d_in[1];
    const float* Win    = (const float*)d_in[2];   // (3D, D)
    const float* bIn    = (const float*)d_in[3];
    const float* Wo     = (const float*)d_in[4];
    const float* bo     = (const float*)d_in[5];
    const float* W1r    = (const float*)d_in[6];
    const float* b1r    = (const float*)d_in[7];
    const float* W1i    = (const float*)d_in[8];
    const float* b1i    = (const float*)d_in[9];
    const float* W2r    = (const float*)d_in[10];
    const float* b2r    = (const float*)d_in[11];
    const float* W2i    = (const float*)d_in[12];
    const float* b2i    = (const float*)d_in[13];
    const float* g1r    = (const float*)d_in[14];
    const float* be1r   = (const float*)d_in[15];
    const float* g1i    = (const float*)d_in[16];
    const float* be1i   = (const float*)d_in[17];
    const float* g2r    = (const float*)d_in[18];
    const float* be2r   = (const float*)d_in[19];
    const float* g2i    = (const float*)d_in[20];
    const float* be2i   = (const float*)d_in[21];

    float* out1 = (float*)d_out;                               // (4,2,T)
    float* out2 = out1 + (size_t)NB * 2 * TLEN;                // (8,BINS,S,2)

    // ---- workspace layout (floats) ----
    const size_t AA   = (size_t)NB * SFR * DM;                 // activation
    const size_t SCSZ = (size_t)NB * NH * SFR * SFR;
    const size_t FF   = (size_t)NB * SFR * DFF;
    float* ws = (float*)d_ws;
    float* xr  = ws;            float* xi  = xr + AA;
    float* qr  = xi + AA;       float* qi  = qr + AA;
    float* kr  = qi + AA;       float* ki  = kr + AA;
    float* vr  = ki + AA;       float* vi  = vr + AA;
    float* ore = vi + AA;       float* oim = ore + AA;
    float* nre = oim + AA;      float* nim = nre + AA;
    float* nr1 = nim + AA;      float* ni1 = nr1 + AA;
    float* nr2 = ni1 + AA;      float* ni2 = nr2 + AA;
    float* sc  = ni2 + AA;
    float* hr  = sc + SCSZ;
    float* hi  = hr + FF;
    float* frames = hr;         // alias: hr dead once ISTFT starts
    float* ct   = hi + FF;
    float* stt  = ct + NFFT;
    float* wsum = stt + NFFT;

    const int MROWS = NB * SFR;                                // 2052
    const long long sBH = (long long)SFR * DM;                 // per-b stride
    const long long sSC = (long long)NH * SFR * SFR;

    // ---- tables + STFT (+pos enc) ----
    tw_kernel<<<(NFFT + 255) / 256, 256, 0, stream>>>(ct, stt);
    wsum_kernel<<<(LOLA + 255) / 256, 256, 0, stream>>>(window, wsum);
    stft_kernel<<<dim3(SFR, 8), 256, 0, stream>>>(mix, window, ct, stt, xr, xi);

    // ---- QKV projections (shared weights across the 8-way batching) ----
    const float* Wq = Win;
    const float* Wk = Win + (size_t)DM * DM;
    const float* Wv = Win + (size_t)2 * DM * DM;
    const float* bq = bIn;  const float* bk = bIn + DM;  const float* bv = bIn + 2 * DM;
    launch_gemm(stream, xr, DM, 0, Wq, DM, 0, 1, qr, DM, 0, MROWS, DM, DM, 1, 1.f, bq, 1.f, 0, 0);
    launch_gemm(stream, xi, DM, 0, Wq, DM, 0, 1, qi, DM, 0, MROWS, DM, DM, 1, 1.f, bq, 1.f, 0, 0);
    launch_gemm(stream, xr, DM, 0, Wk, DM, 0, 1, kr, DM, 0, MROWS, DM, DM, 1, 1.f, bk, 1.f, 0, 0);
    launch_gemm(stream, xi, DM, 0, Wk, DM, 0, 1, ki, DM, 0, MROWS, DM, DM, 1, 1.f, bk, 1.f, 0, 0);
    launch_gemm(stream, xr, DM, 0, Wv, DM, 0, 1, vr, DM, 0, MROWS, DM, DM, 1, 1.f, bv, 1.f, 0, 0);
    launch_gemm(stream, xi, DM, 0, Wv, DM, 0, 1, vi, DM, 0, MROWS, DM, DM, 1, 1.f, bv, 1.f, 0, 0);

    // ---- attention: 4 distinct (q,k) combos; accumulate the 8-way sums ----
    const float scal = 1.0f / sqrtf((float)HD);
    const float* qsrc[4] = { qr, qr, qi, qi };
    const float* ksrc[4] = { kr, ki, kr, ki };
    const float* vRe[4]  = { vr, vi, vi, vr };   // contribution to o_real
    const float sRe[4]   = { +1.f, +1.f, -1.f, -1.f };
    const float* vIm[4]  = { vi, vr, vr, vi };   // contribution to o_imag
    const float sIm[4]   = { +1.f, +1.f, +1.f, -1.f };

    for (int cb = 0; cb < 4; ++cb) {
        for (int h = 0; h < NH; ++h)
            launch_gemm(stream, qsrc[cb] + (size_t)h * HD, DM, sBH,
                        ksrc[cb] + (size_t)h * HD, DM, sBH, 1,
                        sc + (size_t)h * SFR * SFR, SFR, sSC,
                        SFR, SFR, HD, NB, scal, nullptr, 0.f, 0, 0);
        softmax_kernel<<<NB * NH * SFR, 256, 0, stream>>>(sc);
        for (int h = 0; h < NH; ++h) {
            launch_gemm(stream, sc + (size_t)h * SFR * SFR, SFR, sSC,
                        vRe[cb] + (size_t)h * HD, DM, sBH, 0,
                        ore + (size_t)h * HD, DM, sBH,
                        SFR, HD, SFR, NB, sRe[cb], nullptr, 0.f, (cb > 0), 0);
            launch_gemm(stream, sc + (size_t)h * SFR * SFR, SFR, sSC,
                        vIm[cb] + (size_t)h * HD, DM, sBH, 0,
                        oim + (size_t)h * HD, DM, sBH,
                        SFR, HD, SFR, NB, sIm[cb], nullptr, 0.f, (cb > 0), 0);
        }
    }

    // ---- out-projection (bias coeff sums: 0 for real, +2 for imag) ----
    launch_gemm(stream, ore, DM, 0, Wo, DM, 0, 1, nre, DM, 0, MROWS, DM, DM, 1, 1.f, bo, 0.f, 0, 0);
    launch_gemm(stream, oim, DM, 0, Wo, DM, 0, 1, nim, DM, 0, MROWS, DM, DM, 1, 1.f, bo, 2.f, 0, 0);

    // ---- residual + LN1 ----
    ln_kernel<<<MROWS, 256, 0, stream>>>(xr, nre, g1r, be1r, nr1);
    ln_kernel<<<MROWS, 256, 0, stream>>>(xi, nim, g1i, be1i, ni1);

    // ---- FFN layer 1: other_r = relu(nr@W1r+b1r - (ni@W1i+b1i)), other_i = relu(ni@W1r+b1r + nr@W1i+b1i)
    launch_gemm(stream, nr1, DM, 0, W1r, DM, 0, 1, hr, DFF, 0, MROWS, DFF, DM, 1, +1.f, b1r, +1.f, 0, 0);
    launch_gemm(stream, ni1, DM, 0, W1i, DM, 0, 1, hr, DFF, 0, MROWS, DFF, DM, 1, -1.f, b1i, -1.f, 1, 1);
    launch_gemm(stream, ni1, DM, 0, W1r, DM, 0, 1, hi, DFF, 0, MROWS, DFF, DM, 1, +1.f, b1r, +1.f, 0, 0);
    launch_gemm(stream, nr1, DM, 0, W1i, DM, 0, 1, hi, DFF, 0, MROWS, DFF, DM, 1, +1.f, b1i, +1.f, 1, 1);

    // ---- FFN layer 2 with residual (seed with nr/ni, then accumulate) ----
    hipMemcpyAsync(nr2, nr1, AA * sizeof(float), hipMemcpyDeviceToDevice, stream);
    hipMemcpyAsync(ni2, ni1, AA * sizeof(float), hipMemcpyDeviceToDevice, stream);
    launch_gemm(stream, hr, DFF, 0, W2r, DFF, 0, 1, nr2, DM, 0, MROWS, DM, DFF, 1, +1.f, b2r, +1.f, 1, 0);
    launch_gemm(stream, hi, DFF, 0, W2i, DFF, 0, 1, nr2, DM, 0, MROWS, DM, DFF, 1, -1.f, b2i, -1.f, 1, 0);
    launch_gemm(stream, hi, DFF, 0, W2r, DFF, 0, 1, ni2, DM, 0, MROWS, DM, DFF, 1, +1.f, b2r, +1.f, 1, 0);
    launch_gemm(stream, hr, DFF, 0, W2i, DFF, 0, 1, ni2, DM, 0, MROWS, DM, DFF, 1, +1.f, b2i, +1.f, 1, 0);

    // ---- LN2 (in place) ----
    ln_kernel<<<MROWS, 256, 0, stream>>>(nr2, nullptr, g2r, be2r, nr2);
    ln_kernel<<<MROWS, 256, 0, stream>>>(ni2, nullptr, g2i, be2i, ni2);

    // ---- sigmoid gate + zc scatter ----
    gate_scatter_kernel<<<(int)((AA + 255) / 256), 256, 0, stream>>>(xr, xi, nr2, ni2, out2);

    // ---- ISTFT ----
    istft_frames_kernel<<<dim3(SFR, 8), 256, 0, stream>>>(nr2, ni2, window, ct, stt, frames);
    ola_kernel<<<(int)(((size_t)8 * TLEN + 255) / 256), 256, 0, stream>>>(frames, wsum, out1);
}
